// RNN_85306640433621
// MI455X (gfx1250) — matrix operand order repaired and checked by reading, not compiled, not run on hardware
//
#include <hip/hip_runtime.h>
#include <hip/hip_bf16.h>

// RNN scan: h_{t+1} = tanh(x_t @ W1x + h_t @ W1h + b1); out = h_T @ W2 + b2
// BATCH=2048, SEQ=512, HID=256. bf16 WMMA with f32 accumulate on gfx1250.

#define SEQ      512
#define HID      256
#define ROWS     32                // batch rows per workgroup
#define MT       (ROWS / 16)       // M tiles per wave (2)
#define WAVES    16                // one wave per 16 hidden output columns
#define THREADS  (WAVES * 32)
#define TCHUNK   16                // xs staging chunk (steps)
#define HSTRIDE  264               // padded bf16 row stride (bank-conflict-free b128)

typedef __attribute__((ext_vector_type(16))) __bf16 bf16x16;
typedef __attribute__((ext_vector_type(8)))  __bf16 bf16x8;
typedef __attribute__((ext_vector_type(8)))  float  f32x8;

union ATile { bf16x16 v; struct { bf16x8 lo; bf16x8 hi; } s; };

__device__ __forceinline__ float fast_tanh(float x) {
  // tanh(x) = (e^{2x}-1)/(e^{2x}+1), clamped so e never hits inf.
  float xc = __builtin_fmaxf(__builtin_fminf(x, 15.0f), -15.0f);
  float e  = __builtin_amdgcn_exp2f(xc * 2.8853900817779268f);  // exp(2x)
  return (e - 1.0f) * __builtin_amdgcn_rcpf(e + 1.0f);
}

__global__ __launch_bounds__(THREADS)
void rnn_scan_wmma(const float* __restrict__ xs,   // [B, SEQ, 2]
                   const float* __restrict__ W1x,  // [2, HID]
                   const float* __restrict__ W1h,  // [HID, HID]
                   const float* __restrict__ b1,   // [HID]
                   const float* __restrict__ W2,   // [HID, 1]
                   const float* __restrict__ b2,   // [1]
                   float* __restrict__ out)        // [B, 1]
{
  __shared__ __align__(16) __bf16 sh[ROWS * HSTRIDE];          // h_t (bf16)
  __shared__ __align__(16) float  sx[ROWS * TCHUNK * 2];       // xs chunk
  __shared__ float sred[WAVES][ROWS];                          // output partials

  const int tid   = threadIdx.x;
  const int wave  = tid >> 5;
  const int lane  = tid & 31;
  const int l15   = lane & 15;
  const int hsel  = lane >> 4;            // which half of the wave
  const int n     = wave * 16 + l15;      // this lane's hidden column
  const int rowb  = blockIdx.x * ROWS;    // this WG's batch rows

  // ---- zero-init h0 in LDS ----
  {
    unsigned* p = (unsigned*)sh;
    const int ndw = ROWS * HSTRIDE * 2 / 4;
    for (int i = tid; i < ndw; i += THREADS) p[i] = 0u;
  }

  // ---- per-lane constants (column n) ----
  const float b1n = b1[n];
  const float wx0 = W1x[n];
  const float wx1 = W1x[HID + n];
  const float w2n = W2[n];

  // ---- resident B tiles: W1h columns [n], bf16, WMMA B layout ----
  // B tile kt covers K = kt*32 .. +31. Lane layout: n = lane&15;
  // vector elem i holds K = kt*32 + hsel*16 + i.
  bf16x16 B[8];
#pragma unroll
  for (int kt = 0; kt < 8; ++kt) {
    bf16x16 t;
#pragma unroll
    for (int i = 0; i < 16; ++i) {
      const int k = kt * 32 + hsel * 16 + i;
      t[i] = (__bf16)W1h[k * HID + n];
    }
    B[kt] = t;
  }

  f32x8 c[MT];
  __syncthreads();

  for (int t = 0; t < SEQ; ++t) {
    // ---- stage xs chunk [ROWS][TCHUNK][2] every TCHUNK steps ----
    if ((t & (TCHUNK - 1)) == 0) {
      if (tid < ROWS * TCHUNK * 2 / 4) {            // 256 float4 loads
        const int row = tid >> 3;                   // 8 float4 per row
        const int off = tid & 7;
        const size_t g = ((size_t)(rowb + row) * SEQ + t) * 2 + off * 4;
        *(float4*)&sx[row * (TCHUNK * 2) + off * 4] = *(const float4*)&xs[g];
        if (t + TCHUNK < SEQ)
          __builtin_prefetch(&xs[g + TCHUNK * 2], 0, 1);
      }
      __syncthreads();
    }
    const int tt = t & (TCHUNK - 1);

    // ---- C init: b1 + x_t @ W1x (C layout: m = mt*16 + hsel*8 + j) ----
#pragma unroll
    for (int mt = 0; mt < MT; ++mt)
#pragma unroll
      for (int j = 0; j < 8; ++j) {
        const int m = mt * 16 + hsel * 8 + j;
        const float2 xv = *(const float2*)&sx[m * (TCHUNK * 2) + tt * 2];
        c[mt][j] = __builtin_fmaf(xv.x, wx0, __builtin_fmaf(xv.y, wx1, b1n));
      }

    // ---- h_prev @ W1h : 8 K-tiles x MT M-tiles of bf16 WMMA ----
    // A layout: row = mt*16 + (lane&15); elems 0-7 <- K=k0+16*hsel,
    // elems 8-15 <- K=k0+16*hsel+8 (16B contiguous each -> ds_load_b128).
#pragma unroll
    for (int kt = 0; kt < 8; ++kt) {
      const int k0 = kt * 32;
#pragma unroll
      for (int mt = 0; mt < MT; ++mt) {
        ATile a;
        const __bf16* p = &sh[(mt * 16 + l15) * HSTRIDE + k0 + hsel * 16];
        a.s.lo = *(const bf16x8*)p;
        a.s.hi = *(const bf16x8*)(p + 8);
        c[mt] = __builtin_amdgcn_wmma_f32_16x16x32_bf16(
            false, a.v, false, B[kt], (short)0, c[mt], false, false);
      }
    }

    __syncthreads();   // everyone done reading h_prev

    // ---- tanh, keep fp32 in regs, publish bf16 h_next ----
#pragma unroll
    for (int mt = 0; mt < MT; ++mt)
#pragma unroll
      for (int j = 0; j < 8; ++j) {
        const float h = fast_tanh(c[mt][j]);
        c[mt][j] = h;
        const int m = mt * 16 + hsel * 8 + j;
        sh[m * HSTRIDE + n] = (__bf16)h;
      }

    __syncthreads();   // h_next visible
  }

  // ---- out[m] = sum_n h[m][n] * W2[n] + b2 ----
#pragma unroll
  for (int mt = 0; mt < MT; ++mt)
#pragma unroll
    for (int j = 0; j < 8; ++j) {
      float p = c[mt][j] * w2n;
      p += __shfl_xor(p, 1, 32);
      p += __shfl_xor(p, 2, 32);
      p += __shfl_xor(p, 4, 32);
      p += __shfl_xor(p, 8, 32);      // sum over this wave's 16 columns
      if (l15 == 0) sred[wave][mt * 16 + hsel * 8 + j] = p;
    }
  __syncthreads();

  if (tid < ROWS) {
    float acc = b2[0];
#pragma unroll
    for (int w = 0; w < WAVES; ++w) acc += sred[w][tid];
    out[rowb + tid] = acc;
  }
}

extern "C" void kernel_launch(void* const* d_in, const int* in_sizes, int n_in,
                              void* d_out, int out_size, void* d_ws, size_t ws_size,
                              hipStream_t stream) {
  const float* xs  = (const float*)d_in[0];
  const float* W1x = (const float*)d_in[1];
  const float* W1h = (const float*)d_in[2];
  const float* b1  = (const float*)d_in[3];
  const float* W2  = (const float*)d_in[4];
  const float* b2  = (const float*)d_in[5];
  float* out = (float*)d_out;

  const int batch = in_sizes[0] / (SEQ * 2);   // 2048
  const int grid  = batch / ROWS;              // 64 workgroups
  rnn_scan_wmma<<<grid, THREADS, 0, stream>>>(xs, W1x, W1h, b1, W2, b2, out);
}
